// trajectory2seq_48464410968529
// MI455X (gfx1250) — matrix operand order repaired and checked by reading, not compiled, not run on hardware
//
#include <hip/hip_runtime.h>
#include <math.h>

// Problem constants
#define TT   2048
#define BB   64
#define HH   256
#define DD   128
#define WPAD 264            // padded LDS row stride in halfs (breaks 64-bank conflicts)

typedef __attribute__((ext_vector_type(16))) _Float16 v16h;
typedef __attribute__((ext_vector_type(8)))  _Float16 v8h;
typedef __attribute__((ext_vector_type(8)))  float    v8f;

// Native gfx1250 V_TANH_F32 (confirmed emitted) with safe fallback.
__device__ __forceinline__ float fast_tanh(float v) {
#if __has_builtin(__builtin_amdgcn_tanhf)
  return __builtin_amdgcn_tanhf(v);
#elif __has_builtin(__builtin_amdgcn_tanh_f32)
  return __builtin_amdgcn_tanh_f32(v);
#else
  return tanhf(v);
#endif
}

// ---------------------------------------------------------------------------
// WMMA fragment loaders (CDNA5 ISA 7.12.2 layouts, wave32)
// A 16x32 f16: lane&15 = M row; lanes<16 hold K {0..7,16..23}+32*kc, lanes>=16 hold K {8..15,24..31}+32*kc
// B 32x16 f16: lane&15 = N col; lanes<16 hold K {0..7,16..23}+32*kc, lanes>=16 hold K {8..15,24..31}+32*kc
// ---------------------------------------------------------------------------
__device__ __forceinline__ v16h fragA(const _Float16* buf, int row, int stride, int kc, int lane) {
  const _Float16* p = buf + (size_t)row * stride + kc * 32 + ((lane >> 4) << 3);
  v8h lo = *(const v8h*)(p);
  v8h hi = *(const v8h*)(p + 16);
  v16h r;
#pragma unroll
  for (int i = 0; i < 8; ++i) { r[i] = lo[i]; r[8 + i] = hi[i]; }
  return r;
}

__device__ __forceinline__ v16h fragB(const _Float16* buf, int row, int stride, int kc, int lane) {
  const _Float16* p = buf + (size_t)row * stride + kc * 32 + ((lane >> 4) << 3);
  v8h lo = *(const v8h*)(p);
  v8h hi = *(const v8h*)(p + 16);
  v16h r;
#pragma unroll
  for (int i = 0; i < 8; ++i) { r[i] = lo[i]; r[8 + i] = hi[i]; }
  return r;
}

// ---------------------------------------------------------------------------
// K0: convert recurrent/projection weights to f16, fuse bias pairs
// ---------------------------------------------------------------------------
__global__ void __launch_bounds__(256) prep_kernel(
    const float* __restrict__ Whh0, const float* __restrict__ Wih1, const float* __restrict__ Whh1,
    const float* __restrict__ bih0, const float* __restrict__ bhh0,
    const float* __restrict__ bih1, const float* __restrict__ bhh1,
    _Float16* __restrict__ w0, _Float16* __restrict__ w1, _Float16* __restrict__ w2,
    float* __restrict__ bias0, float* __restrict__ bias1) {
  int i = blockIdx.x * 256 + threadIdx.x;
  if (i < 65536)        w0[i]          = (_Float16)Whh0[i];
  else if (i < 131072)  w1[i - 65536]  = (_Float16)Wih1[i - 65536];
  else                  w2[i - 131072] = (_Float16)Whh1[i - 131072];
  if (i < HH)           bias0[i]       = bih0[i] + bhh0[i];
  else if (i < 2 * HH)  bias1[i - HH]  = bih1[i - HH] + bhh1[i - HH];
}

// ---------------------------------------------------------------------------
// K0b: one-hot gather hoisted out of the sequential loop.
// pre0[t,b,n] = W_ih0[n, x[b,t]] + bias0[n] -> d_out used as [T,B,H] scratch.
// ---------------------------------------------------------------------------
__global__ void __launch_bounds__(256) pre0_kernel(
    const int* __restrict__ x, const float* __restrict__ Wih0,
    const float* __restrict__ bias0, float* __restrict__ pre0) {
  const int t = blockIdx.x >> 6;
  const int b = blockIdx.x & 63;
  const int n = threadIdx.x;
  const int xv = x[b * TT + t];                 // uniform per block -> scalar load
  pre0[(size_t)blockIdx.x * HH + n] = Wih0[n * DD + xv] + bias0[n];
}

// ---------------------------------------------------------------------------
// K1: layer-0 recurrence. 1 workgroup, 16 waves (4/SIMD, ~256 VGPR budget:
// no spills); each wave owns 4 n-tiles -> 4 independent WMMA accumulator
// chains for ILP. W_hh0 f16 + h state live in LDS; the previous step's h
// slice is block-copied to global (b128) during the next step's read phase.
// ---------------------------------------------------------------------------
__global__ void __launch_bounds__(512) rnn_rec0_kernel(
    const float* __restrict__ pre0, const _Float16* __restrict__ w0,
    _Float16* __restrict__ h1, float* __restrict__ hn0) {
  extern __shared__ __align__(16) char smem[];
  _Float16* sW = (_Float16*)smem;                             // [256][WPAD]
  _Float16* sh = (_Float16*)(smem + (size_t)HH * WPAD * 2);   // [64][WPAD]

  const int tid  = threadIdx.x;
  const int lane = tid & 31;
  const int wave = tid >> 5;                 // 0..15

  for (int i = tid; i < HH * HH; i += 512) sW[(i >> 8) * WPAD + (i & 255)] = w0[i];
  for (int i = tid; i < BB * WPAD; i += 512) sh[i] = (_Float16)0.f;
  __syncthreads();

  const int mt   = wave & 3;                 // 4 m-tiles (b)
  const int nt0  = (wave >> 2) * 4;          // 4 n-tiles per wave
  const int arow = mt * 16 + (lane & 15);
  const int nloc = lane & 15;
  const int mhi  = (lane >> 4) << 3;         // C/D row group offset
  const int nb   = nt0 * 16 + nloc;          // tiles at nb + {0,16,32,48}

  for (int t = 0; t < TT; ++t) {
    // ---- read phase (LDS reads only; racing nothing before the barrier) ----
    v16h a[8];
#pragma unroll
    for (int kc = 0; kc < 8; ++kc) a[kc] = fragA(sh, arow, WPAD, kc, lane);

    if (t > 0) {                              // block-copy h(t-1): LDS -> global, b128
      _Float16* dst = h1 + (size_t)(t - 1) * (BB * HH);
#pragma unroll
      for (int q = 0; q < 4; ++q) {
        int idx = (tid * 4 + q) * 8;          // 16B chunks
        *(v8h*)(dst + idx) = *(const v8h*)(sh + (idx >> 8) * WPAD + (idx & 255));
      }
    }
    if (t + 1 < TT)                           // next pre slice -> near caches
      __builtin_prefetch(pre0 + (size_t)(t + 1) * BB * HH + tid * 32, 0, 3);
    __syncthreads();

    // ---- compute phase ----
    const float* preT = pre0 + (size_t)t * BB * HH;
    float pv[4][8];                           // pre loads in flight before WMMA chain
#pragma unroll
    for (int r = 0; r < 8; ++r) {
      const int b_ = mt * 16 + mhi + r;
#pragma unroll
      for (int j = 0; j < 4; ++j) pv[j][r] = preT[b_ * HH + nb + 16 * j];
    }

    v8f c[4] = {{}, {}, {}, {}};
#pragma unroll
    for (int kc = 0; kc < 8; ++kc) {          // 4 independent accumulator chains
#pragma unroll
      for (int j = 0; j < 4; ++j) {
        v16h bf = fragB(sW, nb + 16 * j, WPAD, kc, lane);
        c[j] = __builtin_amdgcn_wmma_f32_16x16x32_f16(false, a[kc], false, bf,
                                                      (short)0, c[j], false, false);
      }
    }

#pragma unroll
    for (int r = 0; r < 8; ++r) {
      const int b_ = mt * 16 + mhi + r;
#pragma unroll
      for (int j = 0; j < 4; ++j)
        sh[b_ * WPAD + nb + 16 * j] = (_Float16)fast_tanh(c[j][r] + pv[j][r]);
    }
    __syncthreads();
  }

  // ---- epilogue: final slice + h_n[0] ----
  {
    _Float16* dst = h1 + (size_t)(TT - 1) * (BB * HH);
#pragma unroll
    for (int q = 0; q < 4; ++q) {
      int idx = (tid * 4 + q) * 8;
      *(v8h*)(dst + idx) = *(const v8h*)(sh + (idx >> 8) * WPAD + (idx & 255));
    }
    for (int i = tid; i < BB * HH; i += 512)
      hn0[i] = (float)sh[(i >> 8) * WPAD + (i & 255)];
  }
}

// ---------------------------------------------------------------------------
// K2: layer-1 input projection (big parallel WMMA GEMM).
// pre1[b,t,n] = sum_k h1[t,b,k]*W_ih1[n,k] + bias1[n] -> d_out[b,t,n].
// ---------------------------------------------------------------------------
__global__ void __launch_bounds__(256) proj1_kernel(
    const _Float16* __restrict__ h1, const _Float16* __restrict__ w1,
    const float* __restrict__ bias1, float* __restrict__ outPre) {
  const int tid   = threadIdx.x;
  const int lane  = tid & 31;
  const int wave  = tid >> 5;                       // 0..7
  const int mtile = blockIdx.x * 8 + wave;          // 0..8191
  const int r0    = mtile * 16;

  v16h a[8];
#pragma unroll
  for (int kc = 0; kc < 8; ++kc)
    a[kc] = fragA(h1, r0 + (lane & 15), HH, kc, lane);

  const int nloc = lane & 15;
#pragma unroll 2
  for (int ntp = 0; ntp < 8; ++ntp) {               // two n-tiles per pass
    const int n0 = (ntp * 2) * 16 + nloc;
    const int n1 = n0 + 16;
    v8f c0 = {}, c1 = {};
#pragma unroll
    for (int kc = 0; kc < 8; ++kc) {
      v16h b0 = fragB(w1, n0, HH, kc, lane);
      v16h b1 = fragB(w1, n1, HH, kc, lane);
      c0 = __builtin_amdgcn_wmma_f32_16x16x32_f16(false, a[kc], false, b0,
                                                  (short)0, c0, false, false);
      c1 = __builtin_amdgcn_wmma_f32_16x16x32_f16(false, a[kc], false, b1,
                                                  (short)0, c1, false, false);
    }
    const float bv0 = bias1[n0], bv1 = bias1[n1];
#pragma unroll
    for (int r = 0; r < 8; ++r) {
      int R = r0 + ((lane >> 4) << 3) + r;          // global row = t*64 + b
      int t = R >> 6, b_ = R & 63;
      size_t base = (size_t)b_ * (TT * HH) + (size_t)t * HH;
      outPre[base + n0] = c0[r] + bv0;
      outPre[base + n1] = c1[r] + bv1;
    }
  }
}

// ---------------------------------------------------------------------------
// K3: layer-1 recurrence, in place on d_out (reads pre[b,t,n], writes h2 same addr).
// ---------------------------------------------------------------------------
__global__ void __launch_bounds__(512) rnn_rec1_kernel(
    const _Float16* __restrict__ w2, float* __restrict__ inout,
    float* __restrict__ hn1) {
  extern __shared__ __align__(16) char smem[];
  _Float16* sW = (_Float16*)smem;                             // [256][WPAD]
  _Float16* sh = (_Float16*)(smem + (size_t)HH * WPAD * 2);   // [64][WPAD]

  const int tid  = threadIdx.x;
  const int lane = tid & 31;
  const int wave = tid >> 5;

  for (int i = tid; i < HH * HH; i += 512) sW[(i >> 8) * WPAD + (i & 255)] = w2[i];
  for (int i = tid; i < BB * WPAD; i += 512) sh[i] = (_Float16)0.f;
  __syncthreads();

  const int mt   = wave & 3;
  const int nt0  = (wave >> 2) * 4;
  const int arow = mt * 16 + (lane & 15);
  const int nloc = lane & 15;
  const int mhi  = (lane >> 4) << 3;
  const int nb   = nt0 * 16 + nloc;

  for (int t = 0; t < TT; ++t) {
    v16h a[8];
#pragma unroll
    for (int kc = 0; kc < 8; ++kc) a[kc] = fragA(sh, arow, WPAD, kc, lane);

    if (t + 1 < TT) {                         // next pre column -> near caches (128B/thread)
      const int pb = tid >> 3, pc = tid & 7;
      __builtin_prefetch(inout + ((size_t)pb * TT + (t + 1)) * HH + pc * 32, 0, 3);
    }
    __syncthreads();

    float pv[4][8];                           // early pre loads (overlap WMMA chain)
#pragma unroll
    for (int r = 0; r < 8; ++r) {
      const int b_ = mt * 16 + mhi + r;
      const size_t base = (size_t)b_ * (TT * HH) + (size_t)t * HH;
#pragma unroll
      for (int j = 0; j < 4; ++j) pv[j][r] = inout[base + nb + 16 * j];
    }

    v8f c[4] = {{}, {}, {}, {}};
#pragma unroll
    for (int kc = 0; kc < 8; ++kc) {
#pragma unroll
      for (int j = 0; j < 4; ++j) {
        v16h bf = fragB(sW, nb + 16 * j, WPAD, kc, lane);
        c[j] = __builtin_amdgcn_wmma_f32_16x16x32_f16(false, a[kc], false, bf,
                                                      (short)0, c[j], false, false);
      }
    }

#pragma unroll
    for (int r = 0; r < 8; ++r) {
      const int b_ = mt * 16 + mhi + r;
      const size_t base = (size_t)b_ * (TT * HH) + (size_t)t * HH;
#pragma unroll
      for (int j = 0; j < 4; ++j) {
        float hv = fast_tanh(c[j][r] + pv[j][r]);
        inout[base + nb + 16 * j] = hv;       // overwrite pre with h2 (final output)
        sh[b_ * WPAD + nb + 16 * j] = (_Float16)hv;
      }
    }
    __syncthreads();
  }

  for (int i = tid; i < BB * HH; i += 512)    // h_n[1] epilogue
    hn1[i] = (float)sh[(i >> 8) * WPAD + (i & 255)];
}

// ---------------------------------------------------------------------------
extern "C" void kernel_launch(void* const* d_in, const int* in_sizes, int n_in,
                              void* d_out, int out_size, void* d_ws, size_t ws_size,
                              hipStream_t stream) {
  (void)in_sizes; (void)n_in; (void)out_size; (void)ws_size;
  const int*   x     = (const int*)  d_in[0];
  const float* W_ih0 = (const float*)d_in[1];
  const float* W_hh0 = (const float*)d_in[2];
  const float* b_ih0 = (const float*)d_in[3];
  const float* b_hh0 = (const float*)d_in[4];
  const float* W_ih1 = (const float*)d_in[5];
  const float* W_hh1 = (const float*)d_in[6];
  const float* b_ih1 = (const float*)d_in[7];
  const float* b_hh1 = (const float*)d_in[8];
  float* out = (float*)d_out;

  // workspace layout (bytes)
  char* ws = (char*)d_ws;
  _Float16* w0    = (_Float16*)(ws + 0);        // 128 KB  W_hh0 f16
  _Float16* w1    = (_Float16*)(ws + 131072);   // 128 KB  W_ih1 f16
  _Float16* w2    = (_Float16*)(ws + 262144);   // 128 KB  W_hh1 f16
  float*    bias0 = (float*)   (ws + 393216);   // 1 KB    b_ih0+b_hh0
  float*    bias1 = (float*)   (ws + 394240);   // 1 KB    b_ih1+b_hh1
  _Float16* h1    = (_Float16*)(ws + 395264);   // 64 MB   layer-0 hidden states [T,B,H] f16

  float* hn0 = out + (size_t)BB * TT * HH;      // h_n[0]
  float* hn1 = hn0 + BB * HH;                   // h_n[1]

  const size_t smemRec = (size_t)(HH + BB) * WPAD * 2;  // W + h (~166 KB of 320 KB WGP LDS)

  prep_kernel<<<768, 256, 0, stream>>>(W_hh0, W_ih1, W_hh1, b_ih0, b_hh0, b_ih1, b_hh1,
                                       w0, w1, w2, bias0, bias1);
  // pre0 -> d_out scratch ([T,B,H]); consumed by rec0, then overwritten by proj1
  pre0_kernel<<<TT * BB, 256, 0, stream>>>(x, W_ih0, bias0, out);
  rnn_rec0_kernel<<<1, 512, smemRec, stream>>>(out, w0, h1, hn0);
  proj1_kernel<<<(TT * BB / 16) / 8, 256, 0, stream>>>(h1, w1, bias1, out);
  rnn_rec1_kernel<<<1, 512, smemRec, stream>>>(w2, out, hn1);
}
